// NetConv_8495445311657
// MI455X (gfx1250) — matrix operand order repaired and checked by reading, not compiled, not run on hardware
//
#include <hip/hip_runtime.h>
#include <hip/hip_bf16.h>

typedef __attribute__((ext_vector_type(16))) _Float16 v16h;
typedef __attribute__((ext_vector_type(8)))  float    v8f;

#define WPB   8
#define BLOCK 256

// ---------------------------------------------------------------------------
// WMMA fragment helpers (wave32, v_wmma_f32_16x16x32_f16)
// A: 16(M) x 32(K) f16, row-major in LDS with leading dim ld.
//   lane<16:  elems 0..7 = K 0..7,  8..15 = K 16..23 (M = lane)
//   lane>=16: elems 0..7 = K 8..15, 8..15 = K 24..31 (M = lane-16)
// B: 32(K) x 16(N) f16 taken from transposed weights WT[n][k] (ld = KPAD).
//   lane<16:  elems 0..7 = K 0..7,  8..15 = K 16..23 (N = lane)
//   lane>=16: elems 0..7 = K 8..15, 8..15 = K 24..31 (N = lane-16)
// C/D: 8 x f32; elem r -> M = r + 8*(lane>=16), N = lane&15.
// ---------------------------------------------------------------------------
__device__ __forceinline__ v16h frag_a(const _Float16* X, int ld, int kc, int lane) {
  const int m  = lane & 15;
  const int kb = (lane >> 4) << 3;          // 0 or 8
  const _Float16* p = X + m * ld + kc + kb;
  v16h a;
#pragma unroll
  for (int i = 0; i < 8; ++i) { a[i] = p[i]; a[8 + i] = p[16 + i]; }
  return a;
}

__device__ __forceinline__ v16h frag_b(const _Float16* WT, int ldk, int nt, int kc, int lane) {
  const int n  = (nt << 4) + (lane & 15);
  const int kb = (lane >> 4) << 3;          // 0 or 8
  const _Float16* p = WT + n * ldk + kc + kb;
  v16h b;
#pragma unroll
  for (int i = 0; i < 8; ++i) { b[i] = p[i]; b[8 + i] = p[16 + i]; }
  return b;
}

// Hidden layer: Xout(f16,[16xNOUT]) = leakyrelu(Xin[16xKPAD] @ W + b, 0.2)
template<int KPAD, int NOUT>
__device__ __forceinline__ void layer_h(const _Float16* Xin, _Float16* Xout,
                                        const _Float16* WT, const float* bias, int lane) {
  const int n0 = lane & 15;
  const int hi = (lane >> 4) << 3;
#pragma unroll
  for (int nt = 0; nt < NOUT / 16; ++nt) {
    v8f c = {};
#pragma unroll
    for (int kc = 0; kc < KPAD; kc += 32)
      c = __builtin_amdgcn_wmma_f32_16x16x32_f16(false, frag_a(Xin, KPAD, kc, lane),
                                                 false, frag_b(WT, KPAD, nt, kc, lane),
                                                 (short)0, c, false, false);
    const int n = (nt << 4) + n0;
    const float bv = bias[n];
#pragma unroll
    for (int r = 0; r < 8; ++r) {
      float v = c[r] + bv;
      v = fmaxf(v, 0.2f * v);                // leaky relu, slope 0.2
      Xout[(hi + r) * NOUT + n] = (_Float16)v;
    }
  }
}

// Final layer -> f32 LDS staging tile S[16 x NOUT] (no activation)
template<int KPAD, int NOUT>
__device__ __forceinline__ void layer_f32(const _Float16* Xin, float* S,
                                          const _Float16* WT, const float* bias, int lane) {
  const int n0 = lane & 15;
  const int hi = (lane >> 4) << 3;
#pragma unroll
  for (int nt = 0; nt < NOUT / 16; ++nt) {
    v8f c = {};
#pragma unroll
    for (int kc = 0; kc < KPAD; kc += 32)
      c = __builtin_amdgcn_wmma_f32_16x16x32_f16(false, frag_a(Xin, KPAD, kc, lane),
                                                 false, frag_b(WT, KPAD, nt, kc, lane),
                                                 (short)0, c, false, false);
    const int n = (nt << 4) + n0;
    const float bv = bias[n];
#pragma unroll
    for (int r = 0; r < 8; ++r) S[(hi + r) * NOUT + n] = c[r] + bv;
  }
}

// Final layer -> atomic scatter-add into nfi[dst][NOUT] (segment_sum)
template<int KPAD, int NOUT>
__device__ __forceinline__ void layer_scatter(const _Float16* Xin,
                                              const _Float16* WT, const float* bias,
                                              const int* dstl, float* nfi, int lane) {
  const int n0 = lane & 15;
  const int hi = (lane >> 4) << 3;
#pragma unroll
  for (int nt = 0; nt < NOUT / 16; ++nt) {
    v8f c = {};
#pragma unroll
    for (int kc = 0; kc < KPAD; kc += 32)
      c = __builtin_amdgcn_wmma_f32_16x16x32_f16(false, frag_a(Xin, KPAD, kc, lane),
                                                 false, frag_b(WT, KPAD, nt, kc, lane),
                                                 (short)0, c, false, false);
    const int n = (nt << 4) + n0;
    const float bv = bias[n];
#pragma unroll
    for (int r = 0; r < 8; ++r) {
      const int d = dstl[hi + r];
      if (d >= 0) atomicAdd(nfi + (size_t)d * NOUT + n, c[r] + bv);
    }
  }
}

// Final layer (NOUT=64) -> direct f32 global stores to out[node*64 + n]
template<int KPAD>
__device__ __forceinline__ void layer_out(const _Float16* Xin, float* out,
                                          long base, long nHi,
                                          const _Float16* WT, const float* bias, int lane) {
  const int n0 = lane & 15;
  const int hi = (lane >> 4) << 3;
#pragma unroll
  for (int nt = 0; nt < 4; ++nt) {
    v8f c = {};
#pragma unroll
    for (int kc = 0; kc < KPAD; kc += 32)
      c = __builtin_amdgcn_wmma_f32_16x16x32_f16(false, frag_a(Xin, KPAD, kc, lane),
                                                 false, frag_b(WT, KPAD, nt, kc, lane),
                                                 (short)0, c, false, false);
    const int n = (nt << 4) + n0;
    const float bv = bias[n];
#pragma unroll
    for (int r = 0; r < 8; ++r) {
      const long node = base + hi + r;
      if (node < nHi) out[node * 64 + n] = c[r] + bv;
    }
  }
}

// Convert weights W[Ka x Na] (fp32 row-major) -> transposed f16 WT[NP x KP]
// (zero-padded) + f32 bias in LDS. Block-cooperative.
__device__ __forceinline__ void cvt_wt(const float* W, const float* bias,
                                       int Ka, int Na, int KP, int NP,
                                       _Float16* WT, float* BL, int tid, int nth) {
  for (int i = tid; i < NP * KP; i += nth) {
    const int n = i / KP, k = i - n * KP;
    const float v = (k < Ka && n < Na) ? W[k * Na + n] : 0.0f;
    WT[i] = (_Float16)v;
  }
  for (int n = tid; n < NP; n += nth) BL[n] = (n < Na) ? bias[n] : 0.0f;
}

// Build edge-tile input X[16 x 160] = [nf[src] (64) | nf[dst] (64) | ef (16) | pad 16]
__device__ __forceinline__ void build_edge_x(const float* nf, const float* ef,
                                             const int* src, const int* dst,
                                             long base, long E,
                                             _Float16* X, int* dstl, int lane) {
  const int r = lane & 15;
  const long e = base + r;
  const bool valid = (e < E);
  _Float16* row = X + r * 160;
  if (lane < 16) {
    if (valid) {
      const int s = src[e];
      const float4* p = (const float4*)(nf + (size_t)s * 64);
#pragma unroll
      for (int i = 0; i < 16; ++i) {
        float4 v = p[i];
        _Float16* q = row + i * 4;
        q[0] = (_Float16)v.x; q[1] = (_Float16)v.y; q[2] = (_Float16)v.z; q[3] = (_Float16)v.w;
      }
    } else {
      for (int i = 0; i < 64; ++i) row[i] = (_Float16)0.0f;
    }
  } else {
    if (valid) {
      const int d = dst[e];
      dstl[r] = d;
      const float4* p = (const float4*)(nf + (size_t)d * 64);
#pragma unroll
      for (int i = 0; i < 16; ++i) {
        float4 v = p[i];
        _Float16* q = row + 64 + i * 4;
        q[0] = (_Float16)v.x; q[1] = (_Float16)v.y; q[2] = (_Float16)v.z; q[3] = (_Float16)v.w;
      }
      const float4* pe = (const float4*)(ef + (size_t)e * 16);
#pragma unroll
      for (int i = 0; i < 4; ++i) {
        float4 v = pe[i];
        _Float16* q = row + 128 + i * 4;
        q[0] = (_Float16)v.x; q[1] = (_Float16)v.y; q[2] = (_Float16)v.z; q[3] = (_Float16)v.w;
      }
      for (int i = 144; i < 160; ++i) row[i] = (_Float16)0.0f;
    } else {
      dstl[r] = -1;
      for (int i = 64; i < 160; ++i) row[i] = (_Float16)0.0f;
    }
  }
}

// ---------------------------------------------------------------------------
// Kernel 0: zero workspace + output
// ---------------------------------------------------------------------------
__global__ void k_zero(unsigned* a, size_t na, unsigned* b, size_t nb) {
  const size_t st = (size_t)gridDim.x * blockDim.x;
  for (size_t i = (size_t)blockIdx.x * blockDim.x + threadIdx.x; i < na; i += st) a[i] = 0u;
  for (size_t i = (size_t)blockIdx.x * blockDim.x + threadIdx.x; i < nb; i += st) b[i] = 0u;
}

// ---------------------------------------------------------------------------
// Kernel 1: o2i edge MLP (144->64->64->64->64->128) + segment_sum scatter
// ---------------------------------------------------------------------------
__global__ void __launch_bounds__(BLOCK) k_o2i(
    const float* __restrict__ nf, const float* __restrict__ ef,
    const int* __restrict__ src, const int* __restrict__ dst,
    const float* w0, const float* b0, const float* w1, const float* b1,
    const float* w2, const float* b2, const float* w3, const float* b3,
    const float* w4, const float* b4,
    float* __restrict__ nfi, long E) {
  __shared__ __align__(16) _Float16 WT0[64 * 160];
  __shared__ __align__(16) _Float16 WT1[64 * 64];
  __shared__ __align__(16) _Float16 WT2[64 * 64];
  __shared__ __align__(16) _Float16 WT3[64 * 64];
  __shared__ __align__(16) _Float16 WT4[128 * 64];
  __shared__ float B0[64], B1[64], B2[64], B3[64], B4[128];
  __shared__ __align__(16) _Float16 XA[WPB][16 * 160];
  __shared__ __align__(16) _Float16 XB[WPB][16 * 64];
  __shared__ int DSTL[WPB][16];

  const int tid = threadIdx.x;
  cvt_wt(w0, b0, 144, 64, 160, 64, WT0, B0, tid, BLOCK);
  cvt_wt(w1, b1, 64, 64, 64, 64, WT1, B1, tid, BLOCK);
  cvt_wt(w2, b2, 64, 64, 64, 64, WT2, B2, tid, BLOCK);
  cvt_wt(w3, b3, 64, 64, 64, 64, WT3, B3, tid, BLOCK);
  cvt_wt(w4, b4, 64, 128, 64, 128, WT4, B4, tid, BLOCK);
  __syncthreads();

  const int wv = tid >> 5, lane = tid & 31;
  const long ntiles = (E + 15) >> 4;
  const long wstep = (long)gridDim.x * WPB;
  for (long t = (long)blockIdx.x * WPB + wv; t < ntiles; t += wstep) {
    build_edge_x(nf, ef, src, dst, t << 4, E, XA[wv], DSTL[wv], lane);
    layer_h<160, 64>(XA[wv], XB[wv], WT0, B0, lane);
    layer_h<64, 64>(XB[wv], XA[wv], WT1, B1, lane);
    layer_h<64, 64>(XA[wv], XB[wv], WT2, B2, lane);
    layer_h<64, 64>(XB[wv], XA[wv], WT3, B3, lane);
    layer_scatter<64, 128>(XA[wv], WT4, B4, DSTL[wv], nfi, lane);
  }
}

// ---------------------------------------------------------------------------
// Kernel 2: i2o edge MLP (144->64->64->64->65 pad 80), sigmoid gate,
//           mean-sum scatter (f1), max-key scatter (f2), edge counts.
// ---------------------------------------------------------------------------
__global__ void __launch_bounds__(BLOCK) k_i2o(
    const float* __restrict__ nf, const float* __restrict__ ef,
    const int* __restrict__ src, const int* __restrict__ dst,
    const float* w0, const float* b0, const float* w1, const float* b1,
    const float* w2, const float* b2, const float* w3, const float* b3,
    float* __restrict__ nfo1, unsigned* __restrict__ nfo2k,
    int* __restrict__ cnt, long E) {
  __shared__ __align__(16) _Float16 WT0[64 * 160];
  __shared__ __align__(16) _Float16 WT1[64 * 64];
  __shared__ __align__(16) _Float16 WT2[64 * 64];
  __shared__ __align__(16) _Float16 WT3[80 * 64];
  __shared__ float B0[64], B1[64], B2[64], B3[80];
  __shared__ __align__(16) _Float16 XA[WPB][16 * 160];   // also reused as f32 S[16*80]
  __shared__ __align__(16) _Float16 XB[WPB][16 * 64];
  __shared__ int DSTL[WPB][16];

  const int tid = threadIdx.x;
  cvt_wt(w0, b0, 144, 64, 160, 64, WT0, B0, tid, BLOCK);
  cvt_wt(w1, b1, 64, 64, 64, 64, WT1, B1, tid, BLOCK);
  cvt_wt(w2, b2, 64, 64, 64, 64, WT2, B2, tid, BLOCK);
  cvt_wt(w3, b3, 64, 65, 64, 80, WT3, B3, tid, BLOCK);
  __syncthreads();

  const int wv = tid >> 5, lane = tid & 31;
  const long ntiles = (E + 15) >> 4;
  const long wstep = (long)gridDim.x * WPB;
  for (long t = (long)blockIdx.x * WPB + wv; t < ntiles; t += wstep) {
    build_edge_x(nf, ef, src, dst, t << 4, E, XA[wv], DSTL[wv], lane);
    layer_h<160, 64>(XA[wv], XB[wv], WT0, B0, lane);
    layer_h<64, 64>(XB[wv], XA[wv], WT1, B1, lane);
    layer_h<64, 64>(XA[wv], XB[wv], WT2, B2, lane);
    float* S = (float*)XA[wv];                       // 16*80*4 == 16*160*2 bytes
    layer_f32<64, 80>(XB[wv], S, WT3, B3, lane);

    // epilogue: gate + scatter
    const int m = lane & 15;
    const int half = lane >> 4;
    const int d = DSTL[wv][m];
    if (d >= 0) {
      const float kk = 1.0f / (1.0f + __expf(-S[m * 80 + 0]));
      if (half == 0) {
        atomicAdd(&cnt[d], 1);
#pragma unroll
        for (int j = 0; j < 32; ++j)
          atomicAdd(nfo1 + (size_t)d * 32 + j, S[m * 80 + 1 + j] * kk);
      } else {
#pragma unroll
        for (int j = 0; j < 32; ++j) {
          const float v = S[m * 80 + 33 + j] * kk;
          const unsigned u = __float_as_uint(v);
          const unsigned key = (u & 0x80000000u) ? ~u : (u | 0x80000000u);
          atomicMax(nfo2k + (size_t)d * 32 + j, key);
        }
      }
    }
  }
}

// ---------------------------------------------------------------------------
// Kernel 3: ri node MLP (192->64->64->64->64) on input-mask nodes [0, nHi)
// ---------------------------------------------------------------------------
__global__ void __launch_bounds__(BLOCK) k_ri(
    const float* __restrict__ nf, const float* __restrict__ nfi,
    const float* w0, const float* b0, const float* w1, const float* b1,
    const float* w2, const float* b2, const float* w3, const float* b3,
    float* __restrict__ out, long nHi) {
  __shared__ __align__(16) _Float16 WT0[64 * 192];
  __shared__ __align__(16) _Float16 WT1[64 * 64];
  __shared__ __align__(16) _Float16 WT2[64 * 64];
  __shared__ __align__(16) _Float16 WT3[64 * 64];
  __shared__ float B0[64], B1[64], B2[64], B3[64];
  __shared__ __align__(16) _Float16 XA[WPB][16 * 192];
  __shared__ __align__(16) _Float16 XB[WPB][16 * 64];

  const int tid = threadIdx.x;
  cvt_wt(w0, b0, 192, 64, 192, 64, WT0, B0, tid, BLOCK);
  cvt_wt(w1, b1, 64, 64, 64, 64, WT1, B1, tid, BLOCK);
  cvt_wt(w2, b2, 64, 64, 64, 64, WT2, B2, tid, BLOCK);
  cvt_wt(w3, b3, 64, 64, 64, 64, WT3, B3, tid, BLOCK);
  __syncthreads();

  const int wv = tid >> 5, lane = tid & 31;
  const long ntiles = (nHi + 15) >> 4;
  const long wstep = (long)gridDim.x * WPB;
  for (long t = (long)blockIdx.x * WPB + wv; t < ntiles; t += wstep) {
    const long base = t << 4;
    const int r = lane & 15;
    const long node = base + r;
    const bool valid = (node < nHi);
    _Float16* row = XA[wv] + r * 192;
    if (lane < 16) {
      if (valid) {
        const float4* p = (const float4*)(nf + (size_t)node * 64);
#pragma unroll
        for (int i = 0; i < 16; ++i) {
          float4 v = p[i];
          _Float16* q = row + i * 4;
          q[0] = (_Float16)v.x; q[1] = (_Float16)v.y; q[2] = (_Float16)v.z; q[3] = (_Float16)v.w;
        }
      } else {
        for (int i = 0; i < 64; ++i) row[i] = (_Float16)0.0f;
      }
    } else {
      if (valid) {
        const float4* p = (const float4*)(nfi + (size_t)node * 128);
#pragma unroll
        for (int i = 0; i < 32; ++i) {
          float4 v = p[i];
          _Float16* q = row + 64 + i * 4;
          q[0] = (_Float16)v.x; q[1] = (_Float16)v.y; q[2] = (_Float16)v.z; q[3] = (_Float16)v.w;
        }
      } else {
        for (int i = 64; i < 192; ++i) row[i] = (_Float16)0.0f;
      }
    }
    layer_h<192, 64>(XA[wv], XB[wv], WT0, B0, lane);
    layer_h<64, 64>(XB[wv], XA[wv], WT1, B1, lane);
    layer_h<64, 64>(XA[wv], XB[wv], WT2, B2, lane);
    layer_out<64>(XB[wv], out, base, nHi, WT3, B3, lane);
  }
}

// ---------------------------------------------------------------------------
// Kernel 4: ro node MLP (128->64->64->64->64) on output-mask nodes [nLo, N)
// ---------------------------------------------------------------------------
__global__ void __launch_bounds__(BLOCK) k_ro(
    const float* __restrict__ nf, const float* __restrict__ nfo1,
    const unsigned* __restrict__ nfo2k, const int* __restrict__ cnt,
    const float* w0, const float* b0, const float* w1, const float* b1,
    const float* w2, const float* b2, const float* w3, const float* b3,
    float* __restrict__ out, long nLo, long N) {
  __shared__ __align__(16) _Float16 WT0[64 * 128];
  __shared__ __align__(16) _Float16 WT1[64 * 64];
  __shared__ __align__(16) _Float16 WT2[64 * 64];
  __shared__ __align__(16) _Float16 WT3[64 * 64];
  __shared__ float B0[64], B1[64], B2[64], B3[64];
  __shared__ __align__(16) _Float16 XA[WPB][16 * 128];
  __shared__ __align__(16) _Float16 XB[WPB][16 * 64];

  const int tid = threadIdx.x;
  cvt_wt(w0, b0, 128, 64, 128, 64, WT0, B0, tid, BLOCK);
  cvt_wt(w1, b1, 64, 64, 64, 64, WT1, B1, tid, BLOCK);
  cvt_wt(w2, b2, 64, 64, 64, 64, WT2, B2, tid, BLOCK);
  cvt_wt(w3, b3, 64, 64, 64, 64, WT3, B3, tid, BLOCK);
  __syncthreads();

  const int wv = tid >> 5, lane = tid & 31;
  const long ntiles = (N - nLo + 15) >> 4;
  const long wstep = (long)gridDim.x * WPB;
  for (long t = (long)blockIdx.x * WPB + wv; t < ntiles; t += wstep) {
    const long base = nLo + (t << 4);
    const int r = lane & 15;
    const long node = base + r;
    const bool valid = (node < N);
    _Float16* row = XA[wv] + r * 128;
    if (lane < 16) {
      if (valid) {
        const float4* p = (const float4*)(nf + (size_t)node * 64);
#pragma unroll
        for (int i = 0; i < 16; ++i) {
          float4 v = p[i];
          _Float16* q = row + i * 4;
          q[0] = (_Float16)v.x; q[1] = (_Float16)v.y; q[2] = (_Float16)v.z; q[3] = (_Float16)v.w;
        }
      } else {
        for (int i = 0; i < 64; ++i) row[i] = (_Float16)0.0f;
      }
    } else {
      if (valid) {
        const int c = cnt[node];
        const float inv = 1.0f / fmaxf((float)c, 1.0f);
#pragma unroll
        for (int j = 0; j < 32; ++j)
          row[64 + j] = (_Float16)(nfo1[(size_t)node * 32 + j] * inv);
#pragma unroll
        for (int j = 0; j < 32; ++j) {
          float v = 0.0f;
          if (c > 0) {
            const unsigned key = nfo2k[(size_t)node * 32 + j];
            const unsigned u = (key & 0x80000000u) ? (key ^ 0x80000000u) : ~key;
            v = __uint_as_float(u);
          }
          row[96 + j] = (_Float16)v;
        }
      } else {
        for (int i = 64; i < 128; ++i) row[i] = (_Float16)0.0f;
      }
    }
    layer_h<128, 64>(XA[wv], XB[wv], WT0, B0, lane);
    layer_h<64, 64>(XB[wv], XA[wv], WT1, B1, lane);
    layer_h<64, 64>(XA[wv], XB[wv], WT2, B2, lane);
    layer_out<64>(XB[wv], out, base, N, WT3, B3, lane);
  }
}

// ---------------------------------------------------------------------------
// Host launcher
// ---------------------------------------------------------------------------
extern "C" void kernel_launch(void* const* d_in, const int* in_sizes, int n_in,
                              void* d_out, int out_size, void* d_ws, size_t ws_size,
                              hipStream_t stream) {
  (void)n_in; (void)ws_size;
  const float* nf     = (const float*)d_in[0];
  const float* ef_out = (const float*)d_in[1];
  const float* ef_in  = (const float*)d_in[2];
  const float *ow[5], *ob[5], *iw[4], *ib[4], *rw[4], *rb[4], *qw[4], *qb[4];
  for (int i = 0; i < 5; ++i) { ow[i] = (const float*)d_in[3 + 2 * i];  ob[i] = (const float*)d_in[4 + 2 * i]; }
  for (int i = 0; i < 4; ++i) { iw[i] = (const float*)d_in[13 + 2 * i]; ib[i] = (const float*)d_in[14 + 2 * i]; }
  for (int i = 0; i < 4; ++i) { rw[i] = (const float*)d_in[21 + 2 * i]; rb[i] = (const float*)d_in[22 + 2 * i]; }
  for (int i = 0; i < 4; ++i) { qw[i] = (const float*)d_in[29 + 2 * i]; qb[i] = (const float*)d_in[30 + 2 * i]; }
  const int* src_out = (const int*)d_in[37];
  const int* dst_out = (const int*)d_in[38];
  const int* src_in  = (const int*)d_in[39];
  const int* dst_in  = (const int*)d_in[40];

  const long N = (long)in_sizes[0] / 64;
  const long E = (long)in_sizes[37];
  const long inHi  = (2 * N) / 5;   // input_mask:  idx <  2N/5
  const long outLo = (3 * N) / 5;   // output_mask: idx >= 3N/5

  // workspace layout (all zero-initialized each call)
  float*    nfi   = (float*)d_ws;                         // N*128 f32 segment sums
  float*    nfo1  = nfi + (size_t)N * 128;                // N*32  f32 gated mean sums
  unsigned* nfo2k = (unsigned*)(nfo1 + (size_t)N * 32);   // N*32  u32 monotone max keys
  int*      cnt   = (int*)(nfo2k + (size_t)N * 32);       // N     i32 in-degree counts
  const size_t wsWords = (size_t)N * (128 + 32 + 32 + 1);

  k_zero<<<1024, 256, 0, stream>>>((unsigned*)d_ws, wsWords,
                                   (unsigned*)d_out, (size_t)out_size);

  const long tilesE = (E + 15) / 16;
  long bE = (tilesE + WPB - 1) / WPB; if (bE > 2048) bE = 2048;
  k_o2i<<<(int)bE, BLOCK, 0, stream>>>(nf, ef_out, src_out, dst_out,
      ow[0], ob[0], ow[1], ob[1], ow[2], ob[2], ow[3], ob[3], ow[4], ob[4], nfi, E);
  k_i2o<<<(int)bE, BLOCK, 0, stream>>>(nf, ef_in, src_in, dst_in,
      iw[0], ib[0], iw[1], ib[1], iw[2], ib[2], iw[3], ib[3], nfo1, nfo2k, cnt, E);

  const long tiles3 = (inHi + 15) / 16;
  long b3 = (tiles3 + WPB - 1) / WPB; if (b3 > 2048) b3 = 2048;
  k_ri<<<(int)b3, BLOCK, 0, stream>>>(nf, nfi,
      rw[0], rb[0], rw[1], rb[1], rw[2], rb[2], rw[3], rb[3], (float*)d_out, inHi);

  const long tiles4 = (N - outLo + 15) / 16;
  long b4 = (tiles4 + WPB - 1) / WPB; if (b4 > 2048) b4 = 2048;
  k_ro<<<(int)b4, BLOCK, 0, stream>>>(nf, nfo1, nfo2k, cnt,
      qw[0], qb[0], qw[1], qb[1], qw[2], qb[2], qw[3], qb[3], (float*)d_out, outLo, N);
}
